// LoraAttention_39479339384986
// MI455X (gfx1250) — matrix operand order repaired and checked by reading, not compiled, not run on hardware
//
#include <hip/hip_runtime.h>

// ---------------------------------------------------------------------------
// CDNA5 (gfx1250) LoRA attention. wave32, v_wmma_f32_16x16x32_bf16, async
// HBM->LDS staging (ASYNCcnt), DPP softmax reductions.
// ---------------------------------------------------------------------------

typedef __attribute__((ext_vector_type(16))) __bf16 v16bf;
typedef __attribute__((ext_vector_type(8)))  __bf16 v8bf;
typedef __attribute__((ext_vector_type(8)))  float  v8f;
typedef __attribute__((__vector_size__(4 * sizeof(int)))) int v4i;

#if defined(__has_builtin)
#if __has_builtin(__builtin_amdgcn_global_load_async_to_lds_b128)
#define HAVE_ASYNC_LDS 1
#endif
#endif

// 16B global -> LDS copy (async when the CDNA5 builtin is available).
__device__ __forceinline__ void cp16(const __bf16* g, __bf16* l) {
#ifdef HAVE_ASYNC_LDS
  __builtin_amdgcn_global_load_async_to_lds_b128(
      (__attribute__((address_space(1))) v4i*)(void*)g,
      (__attribute__((address_space(3))) v4i*)(void*)l, 0, 0);
#else
  *reinterpret_cast<v8bf*>(l) = *reinterpret_cast<const v8bf*>(g);
#endif
}

__device__ __forceinline__ void wait_async_n4() {
#ifdef HAVE_ASYNC_LDS
#if __has_builtin(__builtin_amdgcn_s_wait_asynccnt)
  __builtin_amdgcn_s_wait_asynccnt(4);
#else
  asm volatile("s_wait_asynccnt 0x4" ::: "memory");
#endif
#endif
}
__device__ __forceinline__ void wait_async_0() {
#ifdef HAVE_ASYNC_LDS
#if __has_builtin(__builtin_amdgcn_s_wait_asynccnt)
  __builtin_amdgcn_s_wait_asynccnt(0);
#else
  asm volatile("s_wait_asynccnt 0x0" ::: "memory");
#endif
#endif
}

__device__ __forceinline__ v8f wmma_bf16(v16bf a, v16bf b, v8f c) {
  return __builtin_amdgcn_wmma_f32_16x16x32_bf16(
      false, a, false, b, (short)0, c, false, false);
}

// A fragment (16x32 bf16, row-major, leading dim `ld` elements).
__device__ __forceinline__ v16bf ld_a(const __bf16* base, int ld, int lane) {
  const int row = lane & 15;
  const int kh  = (lane >> 4) << 3;  // 0 or 8
  const __bf16* p = base + (size_t)row * ld + kh;
  union { v16bf v; v8bf h[2]; } u;
  u.h[0] = *reinterpret_cast<const v8bf*>(p);       // K = kh .. kh+7
  u.h[1] = *reinterpret_cast<const v8bf*>(p + 16);  // K = 16+kh ..
  return u.v;
}

// B fragment (32x16 bf16): source element [col][k], k contiguous, ld elems.
__device__ __forceinline__ v16bf ld_b(const __bf16* base, int ld, int lane) {
  const int col = lane & 15;
  const int kh  = (lane >> 4) << 3;  // 0 or 8
  const __bf16* p = base + (size_t)col * ld + kh;
  union { v16bf v; v8bf h[2]; } u;
  u.h[0] = *reinterpret_cast<const v8bf*>(p);
  u.h[1] = *reinterpret_cast<const v8bf*>(p + 16);
  return u.v;
}

// --- 16-lane reductions via DPP (stay in the VALU pipe, co-execute w/ WMMA) --
template <int CTRL>
__device__ __forceinline__ float dpp_mov(float x) {
  int v = __builtin_amdgcn_update_dpp(0, __builtin_bit_cast(int, x), CTRL,
                                      0xf, 0xf, true);
  return __builtin_bit_cast(float, v);
}
__device__ __forceinline__ float redmax16(float x) {
  x = fmaxf(x, dpp_mov<0xB1>(x));    // quad_perm [1,0,3,2]  (xor 1)
  x = fmaxf(x, dpp_mov<0x4E>(x));    // quad_perm [2,3,0,1]  (xor 2)
  x = fmaxf(x, dpp_mov<0x141>(x));   // row_half_mirror      (xor 7)
  x = fmaxf(x, dpp_mov<0x140>(x));   // row_mirror           (xor 15)
  return x;
}
__device__ __forceinline__ float redsum16(float x) {
  x += dpp_mov<0xB1>(x);
  x += dpp_mov<0x4E>(x);
  x += dpp_mov<0x141>(x);
  x += dpp_mov<0x140>(x);
  return x;
}

// ---------------------------------------------------------------------------
// fp32 -> bf16 convert
// ---------------------------------------------------------------------------
__global__ void cvt_bf16_kernel(const float* __restrict__ in,
                                __bf16* __restrict__ out, int n) {
  int i = blockIdx.x * blockDim.x + threadIdx.x;
  if (i < n) out[i] = (__bf16)in[i];
}

// ---------------------------------------------------------------------------
// W_eff = W_qkv + W_lora(Q,V) + 0.125 * B@A (Q,V); b_eff = b_lora on Q,V.
// ---------------------------------------------------------------------------
__global__ void prep_weights_kernel(const float* __restrict__ Wq,
                                    const float* __restrict__ Wl,
                                    const float* __restrict__ bl,
                                    const float* __restrict__ Aq,
                                    const float* __restrict__ Bq,
                                    const float* __restrict__ Av,
                                    const float* __restrict__ Bv,
                                    __bf16* __restrict__ Weff,
                                    float* __restrict__ beff) {
  int idx = blockIdx.x * blockDim.x + threadIdx.x;
  if (idx >= 3072 * 1024) return;
  const int o = idx >> 10;
  const int c = idx & 1023;
  const int part = o >> 10;  // 0=Q, 1=K, 2=V
  float w = Wq[idx];
  if (part != 1) {
    w += Wl[idx];
    const float* Bm = (part == 0) ? Bq : Bv;
    const float* Am = (part == 0) ? Aq : Av;
    const int oo = o & 1023;
    float acc = 0.f;
#pragma unroll
    for (int r = 0; r < 8; ++r) acc += Bm[oo * 8 + r] * Am[r * 1024 + c];
    w += 0.125f * acc;  // LORA_SCALE = 1/8
  }
  Weff[idx] = (__bf16)w;
  if (c == 0) beff[o] = (part != 1) ? bl[o] : 0.0f;
}

// ---------------------------------------------------------------------------
// GEMM: out(MxN) = A(MxK) @ W(NxK)^T + bias.   bf16 in, f32 accumulate.
// Block 256 thr = 8 waves (4m x 2n); block tile 128x128, wave tile 32x64.
// K staged in LDS, BK=32, double-buffered via async HBM->LDS.
// ---------------------------------------------------------------------------
template <bool OUT_BF16>
__global__ __launch_bounds__(256) void gemm_bias_kernel(
    const __bf16* __restrict__ A, const __bf16* __restrict__ W,
    const float* __restrict__ bias, void* __restrict__ outp,
    int M, int N, int K) {
  __shared__ __attribute__((aligned(16))) __bf16 la[2][128 * 32];  // 2x8KB
  __shared__ __attribute__((aligned(16))) __bf16 lb[2][128 * 32];  // 2x8KB

  const int tid = threadIdx.x;
  const int lane = tid & 31;
  const int wave = tid >> 5;
  const int wm = wave >> 1;  // 0..3
  const int wn = wave & 1;   // 0..1
  const int m0 = blockIdx.y * 128;
  const int n0 = blockIdx.x * 128;
  (void)M;

  v8f acc[2][4] = {};

  // stage tile (128 rows x 32 k), 2 chunks per thread per matrix
  auto stage = [&](int k0, int buf) {
#pragma unroll
    for (int c = 0; c < 2; ++c) {
      const int i = tid + c * 256;          // 0..511
      const int row = i >> 2;               // 0..127
      const int kc = (i & 3) << 3;          // 0,8,16,24
      cp16(&A[(size_t)(m0 + row) * K + k0 + kc], &la[buf][row * 32 + kc]);
      cp16(&W[(size_t)(n0 + row) * K + k0 + kc], &lb[buf][row * 32 + kc]);
    }
  };

  stage(0, 0);
  int buf = 0;
  for (int k0 = 0; k0 < K; k0 += 32) {
    const bool has_next = (k0 + 32) < K;
    if (has_next) stage(k0 + 32, buf ^ 1);
    if (has_next) wait_async_n4(); else wait_async_0();
    __syncthreads();

    const __bf16* pa = &la[buf][(wm * 32) * 32];
    const __bf16* pb = &lb[buf][(wn * 64) * 32];
    v16bf a0 = ld_a(pa, 32, lane);
    v16bf a1 = ld_a(pa + 16 * 32, 32, lane);
#pragma unroll
    for (int j = 0; j < 4; ++j) {
      v16bf b = ld_b(pb + (j * 16) * 32, 32, lane);
      acc[0][j] = wmma_bf16(a0, b, acc[0][j]);
      acc[1][j] = wmma_bf16(a1, b, acc[1][j]);
    }
    __syncthreads();  // tile consumed before next overwrite
    buf ^= 1;
  }

  const int rbase = (lane >> 4) << 3;
  const int col = lane & 15;
#pragma unroll
  for (int i = 0; i < 2; ++i)
#pragma unroll
    for (int j = 0; j < 4; ++j)
#pragma unroll
      for (int r = 0; r < 8; ++r) {
        const int row = m0 + wm * 32 + i * 16 + rbase + r;
        const int n = n0 + wn * 64 + j * 16 + col;
        float v = acc[i][j][r] + (bias ? bias[n] : 0.0f);
        if (OUT_BF16)
          ((__bf16*)outp)[(size_t)row * N + n] = (__bf16)v;
        else
          ((float*)outp)[(size_t)row * N + n] = v;
      }
}

// ---------------------------------------------------------------------------
// Flash attention over qkv bf16 [B*N, 3072] (o = part*1024 + h*64 + d).
// Grid (N/64, B*H); 128 thr = 4 waves; wave owns 16 Q rows; key tile = 64.
// ---------------------------------------------------------------------------
__global__ __launch_bounds__(128) void attn_kernel(
    const __bf16* __restrict__ qkv, __bf16* __restrict__ attn_out) {
  __shared__ __attribute__((aligned(16))) __bf16 kt[64 * 64];       // [key][feat]
  __shared__ __attribute__((aligned(16))) __bf16 vt[64 * 64];       // [feat][key]
  __shared__ __attribute__((aligned(16))) __bf16 pbuf[4 * 16 * 64];

  const int tid = threadIdx.x;
  const int lane = tid & 31;
  const int wave = tid >> 5;
  const int b = blockIdx.y >> 4;
  const int h = blockIdx.y & 15;
  const int q0 = blockIdx.x * 64 + wave * 16;
  const size_t ld = 3072;

  const __bf16* qb = qkv + (size_t)b * 2048 * ld + h * 64;
  const __bf16* kb = qb + 1024;
  const __bf16* vb = qb + 2048;

  // Q fragments, pre-scaled by D^-0.5 = 1/8 (exact exponent shift in bf16).
  v16bf qa0 = ld_a(qb + (size_t)q0 * ld, ld, lane);
  v16bf qa1 = ld_a(qb + (size_t)q0 * ld + 32, ld, lane);
#pragma unroll
  for (int i = 0; i < 16; ++i) {
    qa0[i] = (__bf16)((float)qa0[i] * 0.125f);
    qa1[i] = (__bf16)((float)qa1[i] * 0.125f);
  }

  v8f o[4] = {};
  float mrow[8], lrow[8];
#pragma unroll
  for (int r = 0; r < 8; ++r) { mrow[r] = -1e30f; lrow[r] = 0.f; }

  const int rbase = (lane >> 4) << 3;
  const int col = lane & 15;
  __bf16* pw = &pbuf[wave * 16 * 64];

  for (int kt0 = 0; kt0 < 2048; kt0 += 64) {
    // Stage K tile (row-major, async path) — 4 chunks of 16B per thread.
    for (int i = tid; i < 64 * 8; i += 128) {
      const int key = i >> 3;
      const int fc = (i & 7) << 3;
      cp16(&kb[(size_t)(kt0 + key) * ld + fc], &kt[key * 64 + fc]);
    }
    // Stage V transposed: vt[feat][key]; vectorized global reads.
    for (int i = tid; i < 64 * 8; i += 128) {
      const int key = i >> 3;
      const int fc = (i & 7) << 3;
      v8bf v = *reinterpret_cast<const v8bf*>(&vb[(size_t)(kt0 + key) * ld + fc]);
#pragma unroll
      for (int e = 0; e < 8; ++e) vt[(fc + e) * 64 + key] = v[e];
    }
    wait_async_0();
    __syncthreads();

    // S(16 x 64 keys) = Q @ K^T, 4 column chunks x 2 k-steps.
    v8f s[4] = {};
#pragma unroll
    for (int j = 0; j < 4; ++j) {
      v16bf bk0 = ld_b(&kt[(j * 16) * 64 + 0], 64, lane);
      v16bf bk1 = ld_b(&kt[(j * 16) * 64 + 32], 64, lane);
      s[j] = wmma_bf16(qa0, bk0, s[j]);
      s[j] = wmma_bf16(qa1, bk1, s[j]);
    }

    // Online softmax; rowwise reduce across 16 column lanes via DPP.
#pragma unroll
    for (int r = 0; r < 8; ++r) {
      float mx = fmaxf(fmaxf(s[0][r], s[1][r]), fmaxf(s[2][r], s[3][r]));
      mx = redmax16(mx);
      const float mnew = fmaxf(mrow[r], mx);
      const float alpha = __expf(mrow[r] - mnew);
      float p[4], ps = 0.f;
#pragma unroll
      for (int j = 0; j < 4; ++j) {
        p[j] = __expf(s[j][r] - mnew);
        ps += p[j];
        pw[(rbase + r) * 64 + j * 16 + col] = (__bf16)p[j];
      }
      ps = redsum16(ps);
      lrow[r] = lrow[r] * alpha + ps;
      mrow[r] = mnew;
#pragma unroll
      for (int j = 0; j < 4; ++j) o[j][r] *= alpha;
    }

    // Wave-private P tile; LDS in-order per wave, drain DScnt before reads.
    asm volatile("s_wait_dscnt 0x0" ::: "memory");
    v16bf pa0 = ld_a(pw, 64, lane);        // keys 0..31
    v16bf pa1 = ld_a(pw + 32, 64, lane);   // keys 32..63
#pragma unroll
    for (int j = 0; j < 4; ++j) {
      v16bf vb0 = ld_b(&vt[(j * 16) * 64 + 0], 64, lane);
      v16bf vb1 = ld_b(&vt[(j * 16) * 64 + 32], 64, lane);
      o[j] = wmma_bf16(pa0, vb0, o[j]);
      o[j] = wmma_bf16(pa1, vb1, o[j]);
    }
    __syncthreads();  // before next tile overwrites kt/vt
  }

  // Normalize, write (B*N, H*D) bf16.
#pragma unroll
  for (int r = 0; r < 8; ++r) {
    const float inv = 1.0f / lrow[r];
    const size_t orow = (size_t)b * 2048 + q0 + rbase + r;
#pragma unroll
    for (int j = 0; j < 4; ++j)
      attn_out[orow * 1024 + h * 64 + j * 16 + col] = (__bf16)(o[j][r] * inv);
  }
}

// ---------------------------------------------------------------------------
// Launch
// ---------------------------------------------------------------------------
extern "C" void kernel_launch(void* const* d_in, const int* in_sizes, int n_in,
                              void* d_out, int out_size, void* d_ws,
                              size_t ws_size, hipStream_t stream) {
  (void)in_sizes; (void)n_in; (void)out_size; (void)ws_size;
  const float* x      = (const float*)d_in[0];
  const float* W_qkv  = (const float*)d_in[1];
  const float* W_lora = (const float*)d_in[2];
  const float* b_lora = (const float*)d_in[3];
  const float* A_q    = (const float*)d_in[4];
  const float* B_q    = (const float*)d_in[5];
  const float* A_v    = (const float*)d_in[6];
  const float* B_v    = (const float*)d_in[7];
  const float* W_out  = (const float*)d_in[8];
  const float* b_out  = (const float*)d_in[9];
  float* out = (float*)d_out;

  char* ws = (char*)d_ws;
  size_t off = 0;
  auto alloc = [&](size_t bytes) -> void* {
    void* p = ws + off;
    off = (off + bytes + 255) & ~(size_t)255;
    return p;
  };
  __bf16* x_bf    = (__bf16*)alloc((size_t)8192 * 1024 * 2);
  __bf16* W_eff   = (__bf16*)alloc((size_t)3072 * 1024 * 2);
  float*  b_eff   = (float*) alloc((size_t)3072 * 4);
  __bf16* qkv     = (__bf16*)alloc((size_t)8192 * 3072 * 2);
  __bf16* attn    = (__bf16*)alloc((size_t)8192 * 1024 * 2);
  __bf16* Wout_bf = (__bf16*)alloc((size_t)1024 * 1024 * 2);

  cvt_bf16_kernel<<<(8192 * 1024 + 255) / 256, 256, 0, stream>>>(x, x_bf,
                                                                 8192 * 1024);
  cvt_bf16_kernel<<<(1024 * 1024 + 255) / 256, 256, 0, stream>>>(
      W_out, Wout_bf, 1024 * 1024);
  prep_weights_kernel<<<(3072 * 1024 + 255) / 256, 256, 0, stream>>>(
      W_qkv, W_lora, b_lora, A_q, B_q, A_v, B_v, W_eff, b_eff);

  // qkv = x @ W_eff^T + b_eff  (8192 x 3072 x 1024), bf16 out
  gemm_bias_kernel<true><<<dim3(3072 / 128, 8192 / 128), 256, 0, stream>>>(
      x_bf, W_eff, b_eff, (void*)qkv, 8192, 3072, 1024);

  // flash attention
  attn_kernel<<<dim3(2048 / 64, 4 * 16), 128, 0, stream>>>(qkv, attn);

  // out = attn @ W_out^T + b_out  (8192 x 1024 x 1024), f32 out
  gemm_bias_kernel<false><<<dim3(1024 / 128, 8192 / 128), 256, 0, stream>>>(
      attn, Wout_bf, b_out, (void*)out, 8192, 1024, 1024);
}
